// Arch20GraphEncoder_35639638622215
// MI455X (gfx1250) — matrix operand order repaired and checked by reading, not compiled, not run on hardware
//
#include <hip/hip_runtime.h>
#include <hip/hip_bf16.h>
#include <math.h>

#define HDIM 128

typedef __attribute__((ext_vector_type(16))) __bf16 v16bf;
typedef __attribute__((ext_vector_type(8)))  float  v8f;

// Non-returning f32 global atomic add via CDNA5 ISA (avoids any CAS-loop fallback).
// Tracked with STOREcnt; completion guaranteed at kernel end (implicit wait at s_endpgm),
// and all consumers run in later kernel launches on the same stream.
__device__ __forceinline__ void atom_add_f32(float* p, float v) {
    asm volatile("global_atomic_add_f32 %0, %1, off" : : "v"(p), "v"(v) : "memory");
}

// ---------------------------------------------------------------------------
// Weight pack: f32 [128(K),128(N)] row-major -> WMMA B-operand bf16 layout.
// For kb in 0..3 (K blocks of 32), nt in 0..7 (N tiles of 16), lane in 0..31,
// idx in 0..15: element = W[kb*32 + 16*(lane>>4) + idx][nt*16 + (lane&15)]
// so each lane's 16-element fragment is 32 contiguous bytes (two b128 loads).
// ---------------------------------------------------------------------------
__global__ void pack_w_kernel(const float* __restrict__ W, __bf16* __restrict__ out) {
    int tid  = blockIdx.x * blockDim.x + threadIdx.x;   // 0..16383
    int idx  = tid & 15;
    int lane = (tid >> 4) & 31;
    int nt   = (tid >> 9) & 7;
    int kb   = (tid >> 12) & 3;
    int half = lane >> 4;
    int n    = nt * 16 + (lane & 15);
    int k    = kb * 32 + 16 * half + idx;               // B lane mapping: 2 K per vgpr pair
    out[tid] = (__bf16)W[k * HDIM + n];
}

// rwse_proj[n,c] = relu(sum_j rwse[n,j]*w[j,c] + b[c])
__global__ void rwse_proj_kernel(const float* __restrict__ rwse, const float* __restrict__ w,
                                 const float* __restrict__ b, float* __restrict__ out,
                                 int Nn, int RW) {
    int i = blockIdx.x * blockDim.x + threadIdx.x;
    if (i >= Nn * HDIM) return;
    int n = i / HDIM, c = i % HDIM;
    float acc = b[c];
    for (int j = 0; j < RW; ++j) acc += rwse[n * RW + j] * w[j * HDIM + c];
    out[i] = fmaxf(acc, 0.f);
}

// h[p,:] = (atom_emb[x_ids[p]] + rwse_proj[node_ids[p]]) * valid[p]   (float4 per thread)
__global__ void init_h_kernel(const float* __restrict__ atom, const float* __restrict__ rp,
                              const int* __restrict__ xids, const int* __restrict__ nids,
                              const int* __restrict__ valid, float* __restrict__ h, int SK) {
    int t = blockIdx.x * blockDim.x + threadIdx.x;
    int p = t >> 5;
    int c4 = (t & 31) * 4;
    if (p >= SK) return;
    float vf = (float)valid[p];
    float4 a = *(const float4*)(atom + (size_t)xids[p] * HDIM + c4);
    float4 r = *(const float4*)(rp + (size_t)nids[p] * HDIM + c4);
    float4 o = { (a.x + r.x) * vf, (a.y + r.y) * vf, (a.z + r.z) * vf, (a.w + r.w) * vf };
    *(float4*)(h + (size_t)p * HDIM + c4) = o;
}

// One wave per edge: agg[dst] += relu(h[src] + bond_emb[eid]); float4 per lane.
// This is the HBM-roofline kernel (Ei * 1 KB per layer) -> single-instruction
// GLOBAL_ATOMIC_ADD_F32 scatter, coalesced b128 gathers.
__global__ void edge_scatter_kernel(const float* __restrict__ h, const int* __restrict__ ei,
                                    const float* __restrict__ bond, const int* __restrict__ eids,
                                    float* __restrict__ agg, int Ei) {
    int gid = blockIdx.x * blockDim.x + threadIdx.x;
    int e = gid >> 5, lane = gid & 31;
    if (e >= Ei) return;
    int src = ei[e], dst = ei[Ei + e];
    float4 a = *(const float4*)(h + (size_t)src * HDIM + lane * 4);
    float4 b = *(const float4*)(bond + (size_t)eids[e] * HDIM + lane * 4);
    float* d = agg + (size_t)dst * HDIM + lane * 4;
    atom_add_f32(d + 0, fmaxf(a.x + b.x, 0.f));
    atom_add_f32(d + 1, fmaxf(a.y + b.y, 0.f));
    atom_add_f32(d + 2, fmaxf(a.z + b.z, 0.f));
    atom_add_f32(d + 3, fmaxf(a.w + b.w, 0.f));
}

// out_bf16 = bf16((1+eps)*h + agg)
__global__ void combine_bf_kernel(const float* __restrict__ h, const float* __restrict__ agg,
                                  const float* __restrict__ epsP, __bf16* __restrict__ out,
                                  size_t n4) {
    size_t t = (size_t)blockIdx.x * blockDim.x + threadIdx.x;
    if (t >= n4) return;
    float e1 = 1.f + *epsP;
    size_t i = t * 4;
    float4 hv = *(const float4*)(h + i);
    float4 av = *(const float4*)(agg + i);
    __bf16 o[4];
    o[0] = (__bf16)(e1 * hv.x + av.x);
    o[1] = (__bf16)(e1 * hv.y + av.y);
    o[2] = (__bf16)(e1 * hv.z + av.z);
    o[3] = (__bf16)(e1 * hv.w + av.w);
    *(uint2*)(out + i) = *(uint2*)o;
}

// ---------------------------------------------------------------------------
// WMMA GEMM: out[M,128] = A[M,128](bf16, row-major) @ Bpacked[128,128] (+epilogue)
// Block = 8 waves; wave w covers N tile w for one 16-row M tile; K loop = 4 WMMAs.
// ---------------------------------------------------------------------------
enum { EPI_RELU_BF16 = 0, EPI_RAW_F32 = 1, EPI_BF16 = 2, EPI_GELU_Q_BF16 = 3 };

template <int EPI>
__global__ __launch_bounds__(256)
void gemm128_kernel(const __bf16* __restrict__ A, const __bf16* __restrict__ Bp,
                    const float* __restrict__ bias, const float* __restrict__ q,
                    float* __restrict__ outF, __bf16* __restrict__ outB, int qdiv) {
    int tileM = blockIdx.x;
    int w = threadIdx.x >> 5;
    int lane = threadIdx.x & 31;
    int half = lane >> 4, r = lane & 15;
    const __bf16* arow = A + ((size_t)tileM * 16 + r) * HDIM;
    v8f acc = {};
#pragma unroll
    for (int kb = 0; kb < 4; ++kb) {
        union { v16bf v; uint4 u[2]; } a, b;
        a.u[0] = *(const uint4*)(arow + kb * 32 + 16 * half);
        a.u[1] = *(const uint4*)(arow + kb * 32 + 16 * half + 8);
        const __bf16* bp = Bp + (((size_t)kb * 8 + w) * 32 + lane) * 16;
        b.u[0] = *(const uint4*)(bp);
        b.u[1] = *(const uint4*)(bp + 8);
        acc = __builtin_amdgcn_wmma_f32_16x16x32_bf16(false, a.v, false, b.v,
                                                      (short)0, acc, false, false);
    }
    int col = w * 16 + r;
    float bs = 0.f;
    if (EPI == EPI_RELU_BF16 || EPI == EPI_RAW_F32) bs = bias[col];
    float qv = 0.f;
    if (EPI == EPI_GELU_Q_BF16) qv = q[(size_t)((tileM * 16) / qdiv) * HDIM + col];
#pragma unroll
    for (int i = 0; i < 8; ++i) {
        size_t orow = (size_t)tileM * 16 + i + 8 * half;
        float x = acc[i] + bs;
        if (EPI == EPI_RELU_BF16) {
            outB[orow * HDIM + col] = (__bf16)fmaxf(x, 0.f);
        } else if (EPI == EPI_RAW_F32) {
            outF[orow * HDIM + col] = x;
        } else if (EPI == EPI_BF16) {
            outB[orow * HDIM + col] = (__bf16)x;
        } else {
            float y = x + qv;
            float g = 0.5f * y * (1.f + erff(y * 0.70710678118f));
            outB[orow * HDIM + col] = (__bf16)g;
        }
    }
}

// Column-wise sum / sumsq of relu(z) over a strip of 64 rows -> atomics into stats[256]
__global__ void bn_stats_kernel(const float* __restrict__ z, float* __restrict__ stats) {
    int c = threadIdx.x;
    int r0 = blockIdx.x * 64;
    float s = 0.f, ss = 0.f;
    for (int j = 0; j < 64; ++j) {
        float x = fmaxf(z[(size_t)(r0 + j) * HDIM + c], 0.f);
        s += x; ss += x * x;
    }
    atom_add_f32(&stats[c], s);
    atom_add_f32(&stats[HDIM + c], ss);
}

// out_bf16 = bf16( resid + (relu(z)-mu)*rsqrt(var+1e-5)*g + b )   (resid optional)
__global__ void bn_apply_kernel(const float* __restrict__ z, const float* __restrict__ stats,
                                const float* __restrict__ g, const float* __restrict__ b,
                                const float* __restrict__ resid, __bf16* __restrict__ outB,
                                float invM, size_t n4) {
    size_t t = (size_t)blockIdx.x * blockDim.x + threadIdx.x;
    if (t >= n4) return;
    size_t i = t * 4;
    int c = (int)(i % HDIM);
    float4 zv = *(const float4*)(z + i);
    float zz[4] = { zv.x, zv.y, zv.z, zv.w };
    __bf16 o[4];
#pragma unroll
    for (int j = 0; j < 4; ++j) {
        float mu = stats[c + j] * invM;
        float var = stats[HDIM + c + j] * invM - mu * mu;
        float x = fmaxf(zz[j], 0.f);
        float y = (x - mu) * rsqrtf(var + 1e-5f) * g[c + j] + b[c + j];
        if (resid) y += resid[i + j];
        o[j] = (__bf16)y;
    }
    *(uint2*)(outB + i) = *(uint2*)o;
}

// h_node[n] = mean over m subgraph roots (rows (n*m+s)*k of h)
__global__ void node_pool_kernel(const float* __restrict__ h, float* __restrict__ hn,
                                 int Nn, int m, int k) {
    int t = blockIdx.x * blockDim.x + threadIdx.x;
    int n = t >> 5;
    int c4 = (t & 31) * 4;
    if (n >= Nn) return;
    float4 acc = { 0, 0, 0, 0 };
    for (int s = 0; s < m; ++s) {
        float4 v = *(const float4*)(h + ((size_t)(n * m + s) * k) * HDIM + c4);
        acc.x += v.x; acc.y += v.y; acc.z += v.z; acc.w += v.w;
    }
    float inv = 1.f / (float)m;
    float4 o = { acc.x * inv, acc.y * inv, acc.z * inv, acc.w * inv };
    *(float4*)(hn + (size_t)n * HDIM + c4) = o;
}

// One wave per row: h = (h + LN(z3)*g + b) * valid  (wave32 butterfly reduce)
__global__ void ln_res_kernel(const float* __restrict__ z3, const float* __restrict__ g,
                              const float* __restrict__ b, const int* __restrict__ valid,
                              float* __restrict__ h, int SK) {
    int gid = blockIdx.x * blockDim.x + threadIdx.x;
    int p = gid >> 5, lane = gid & 31;
    if (p >= SK) return;
    float4 zv = *(const float4*)(z3 + (size_t)p * HDIM + lane * 4);
    float s = zv.x + zv.y + zv.z + zv.w;
    float ss = zv.x * zv.x + zv.y * zv.y + zv.z * zv.z + zv.w * zv.w;
    for (int o = 16; o; o >>= 1) {
        s  += __shfl_xor(s, o, 32);
        ss += __shfl_xor(ss, o, 32);
    }
    float mu = s * (1.f / HDIM);
    float var = ss * (1.f / HDIM) - mu * mu;
    float rs = rsqrtf(var + 1e-5f);
    float vf = (float)valid[p];
    float4 hv = *(const float4*)(h + (size_t)p * HDIM + lane * 4);
    float4 gv = *(const float4*)(g + lane * 4);
    float4 bv = *(const float4*)(b + lane * 4);
    float4 o4;
    o4.x = (hv.x + ((zv.x - mu) * rs * gv.x + bv.x)) * vf;
    o4.y = (hv.y + ((zv.y - mu) * rs * gv.y + bv.y)) * vf;
    o4.z = (hv.z + ((zv.z - mu) * rs * gv.z + bv.z)) * vf;
    o4.w = (hv.w + ((zv.w - mu) * rs * gv.w + bv.w)) * vf;
    *(float4*)(h + (size_t)p * HDIM + lane * 4) = o4;
}

// Final: per-subgraph masked mean -> node mean -> segment_sum into out[NG,128]
__global__ void pool_out_kernel(const float* __restrict__ h, const int* __restrict__ valid,
                                const int* __restrict__ batch, float* __restrict__ out,
                                int m, int k) {
    int n = blockIdx.x;
    int c = threadIdx.x;            // 128 threads = cols
    __shared__ float cnt[16];
    if (c < m) {
        int s = n * m + c;
        float cc = 0.f;
        for (int j = 0; j < k; ++j) cc += (float)valid[s * k + j];
        cnt[c] = fmaxf(cc, 1.f);
    }
    __syncthreads();
    float acc = 0.f;
    for (int s = 0; s < m; ++s) {
        float sum = 0.f;
        const float* base = h + ((size_t)(n * m + s) * k) * HDIM + c;
        for (int j = 0; j < k; ++j) sum += base[(size_t)j * HDIM];
        acc += sum / cnt[s];
    }
    atom_add_f32(&out[(size_t)batch[n] * HDIM + c], acc / (float)m);
}

// ---------------------------------------------------------------------------
extern "C" void kernel_launch(void* const* d_in, const int* in_sizes, int n_in,
                              void* d_out, int out_size, void* d_ws, size_t ws_size,
                              hipStream_t stream) {
    (void)n_in; (void)ws_size;
    const float* atom    = (const float*)d_in[0];
    const float* bond    = (const float*)d_in[1];
    const float* rwse_w  = (const float*)d_in[2];
    const float* rwse_b  = (const float*)d_in[3];
    const float* rwse    = (const float*)d_in[4];
    const float* l_eps   = (const float*)d_in[5];
    const float* l_w1    = (const float*)d_in[6];
    const float* l_b1    = (const float*)d_in[7];
    const float* l_w2    = (const float*)d_in[8];
    const float* l_b2    = (const float*)d_in[9];
    const float* l_bng   = (const float*)d_in[10];
    const float* l_bnb   = (const float*)d_in[11];
    const float* g_eps   = (const float*)d_in[12];
    const float* g_w1    = (const float*)d_in[13];
    const float* g_b1    = (const float*)d_in[14];
    const float* g_w2    = (const float*)d_in[15];
    const float* g_b2    = (const float*)d_in[16];
    const float* g_bng   = (const float*)d_in[17];
    const float* g_bnb   = (const float*)d_in[18];
    const float* bcast_w = (const float*)d_in[19];
    const float* cat_w1  = (const float*)d_in[20];
    const float* cat_b1  = (const float*)d_in[21];
    const float* cat_w2  = (const float*)d_in[22];
    const float* cat_b2  = (const float*)d_in[23];
    const float* ln_g    = (const float*)d_in[24];
    const float* ln_b    = (const float*)d_in[25];
    const int* x_ids     = (const int*)d_in[26];
    const int* intra_ei  = (const int*)d_in[27];
    const int* intra_ea  = (const int*)d_in[28];
    const int* global_ei = (const int*)d_in[29];
    const int* global_ea = (const int*)d_in[30];
    const int* node_ids  = (const int*)d_in[31];
    const int* valid     = (const int*)d_in[32];
    const int* batch     = (const int*)d_in[33];

    const int SK = in_sizes[26];
    const int Ei = in_sizes[28];
    const int Eg = in_sizes[30];
    const int L  = in_sizes[5];
    const int RW = in_sizes[2] / HDIM;
    const int Nn = in_sizes[4] / RW;
    const int kk = 32;                 // setup constant
    const int S  = SK / kk;
    const int m  = S / Nn;
    const int qdiv = kk * m;           // flat rows per node (64)

    // ---- workspace carve (256B aligned) ----
    char* ws = (char*)d_ws;
    auto carve = [&](size_t bytes) -> char* {
        char* p = ws;
        ws += (bytes + 255) & ~(size_t)255;
        return p;
    };
    float*  h     = (float*) carve((size_t)SK * HDIM * 4);
    float*  zbuf  = (float*) carve((size_t)SK * HDIM * 4);   // agg -> z -> z3
    __bf16* bfA   = (__bf16*)carve((size_t)SK * HDIM * 2);   // hh -> h_local
    __bf16* bfB   = (__bf16*)carve((size_t)SK * HDIM * 2);   // hid -> hidden
    __bf16* wpk   = (__bf16*)carve((size_t)8 * L * 16384 * 2);
    float*  rp    = (float*) carve((size_t)Nn * HDIM * 4);
    float*  hn    = (float*) carve((size_t)Nn * HDIM * 4);
    float*  nagg  = (float*) carve((size_t)Nn * HDIM * 4);
    float*  nz    = (float*) carve((size_t)Nn * HDIM * 4);
    float*  qbuf  = (float*) carve((size_t)Nn * HDIM * 4);
    __bf16* nbfA  = (__bf16*)carve((size_t)Nn * HDIM * 2);
    __bf16* nbfB  = (__bf16*)carve((size_t)Nn * HDIM * 2);
    __bf16* hn2bf = (__bf16*)carve((size_t)Nn * HDIM * 2);
    __bf16* hpjbf = (__bf16*)carve((size_t)Nn * HDIM * 2);
    float*  stats = (float*) carve(256 * 4);
    float*  nstats= (float*) carve(256 * 4);

    const size_t HH = (size_t)HDIM * HDIM;

    // ---- pack all weight matrices into WMMA B layout (bf16) ----
    for (int i = 0; i < L; ++i) {
        const float* mats[8] = {
            l_w1 + (size_t)i * HH, l_w2 + (size_t)i * HH,
            g_w1 + (size_t)i * HH, g_w2 + (size_t)i * HH,
            bcast_w + (size_t)i * HH,
            cat_w1 + (size_t)i * 2 * HH,            // top half  (h_local path)
            cat_w1 + (size_t)i * 2 * HH + HH,       // bottom half (broadcast path)
            cat_w2 + (size_t)i * HH
        };
        for (int jm = 0; jm < 8; ++jm)
            pack_w_kernel<<<64, 256, 0, stream>>>(mats[jm], wpk + ((size_t)i * 8 + jm) * 16384);
    }

    rwse_proj_kernel<<<(Nn * HDIM + 255) / 256, 256, 0, stream>>>(rwse, rwse_w, rwse_b, rp, Nn, RW);
    init_h_kernel<<<(SK * 32 + 255) / 256, 256, 0, stream>>>(atom, rp, x_ids, node_ids, valid, h, SK);

    const size_t hbytes = (size_t)SK * HDIM * 4;
    const size_t skn4 = (size_t)SK * HDIM / 4;
    const size_t nnn4 = (size_t)Nn * HDIM / 4;

    for (int i = 0; i < L; ++i) {
        const __bf16* Wl1  = wpk + ((size_t)i * 8 + 0) * 16384;
        const __bf16* Wl2  = wpk + ((size_t)i * 8 + 1) * 16384;
        const __bf16* Wg1  = wpk + ((size_t)i * 8 + 2) * 16384;
        const __bf16* Wg2  = wpk + ((size_t)i * 8 + 3) * 16384;
        const __bf16* Wbc  = wpk + ((size_t)i * 8 + 4) * 16384;
        const __bf16* Wc1t = wpk + ((size_t)i * 8 + 5) * 16384;
        const __bf16* Wc1b = wpk + ((size_t)i * 8 + 6) * 16384;
        const __bf16* Wc2  = wpk + ((size_t)i * 8 + 7) * 16384;

        // ---- local GINE ----
        hipMemsetAsync(zbuf, 0, hbytes, stream);
        edge_scatter_kernel<<<Ei * 32 / 256, 256, 0, stream>>>(h, intra_ei, bond, intra_ea, zbuf, Ei);
        combine_bf_kernel<<<(int)((skn4 + 255) / 256), 256, 0, stream>>>(h, zbuf, l_eps + i, bfA, skn4);
        gemm128_kernel<EPI_RELU_BF16><<<SK / 16, 256, 0, stream>>>(bfA, Wl1, l_b1 + i * HDIM, nullptr, nullptr, bfB, 1);
        gemm128_kernel<EPI_RAW_F32><<<SK / 16, 256, 0, stream>>>(bfB, Wl2, l_b2 + i * HDIM, nullptr, zbuf, nullptr, 1);
        hipMemsetAsync(stats, 0, 256 * 4, stream);
        bn_stats_kernel<<<SK / 64, HDIM, 0, stream>>>(zbuf, stats);
        bn_apply_kernel<<<(int)((skn4 + 255) / 256), 256, 0, stream>>>(
            zbuf, stats, l_bng + i * HDIM, l_bnb + i * HDIM, nullptr, bfA, 1.f / SK, skn4);

        // ---- node-level GINE ----
        node_pool_kernel<<<(Nn * 32 + 255) / 256, 256, 0, stream>>>(h, hn, Nn, m, kk);
        hipMemsetAsync(nagg, 0, (size_t)Nn * HDIM * 4, stream);
        edge_scatter_kernel<<<Eg * 32 / 256, 256, 0, stream>>>(hn, global_ei, bond, global_ea, nagg, Eg);
        combine_bf_kernel<<<(int)((nnn4 + 255) / 256), 256, 0, stream>>>(hn, nagg, g_eps + i, nbfA, nnn4);
        gemm128_kernel<EPI_RELU_BF16><<<Nn / 16, 256, 0, stream>>>(nbfA, Wg1, g_b1 + i * HDIM, nullptr, nullptr, nbfB, 1);
        gemm128_kernel<EPI_RAW_F32><<<Nn / 16, 256, 0, stream>>>(nbfB, Wg2, g_b2 + i * HDIM, nullptr, nz, nullptr, 1);
        hipMemsetAsync(nstats, 0, 256 * 4, stream);
        bn_stats_kernel<<<Nn / 64, HDIM, 0, stream>>>(nz, nstats);
        bn_apply_kernel<<<(int)((nnn4 + 255) / 256), 256, 0, stream>>>(
            nz, nstats, g_bng + i * HDIM, g_bnb + i * HDIM, hn, hn2bf, 1.f / Nn, nnn4);

        // ---- broadcast path: q = (h_node2 @ bcast) @ cat_w1_bot + cat_b1 ----
        gemm128_kernel<EPI_BF16><<<Nn / 16, 256, 0, stream>>>(hn2bf, Wbc, nullptr, nullptr, nullptr, hpjbf, 1);
        gemm128_kernel<EPI_RAW_F32><<<Nn / 16, 256, 0, stream>>>(hpjbf, Wc1b, cat_b1 + i * HDIM, nullptr, qbuf, nullptr, 1);

        // ---- cat MLP: hidden = gelu(h_local @ W1top + q[row/64]); z3 = hidden @ W2 + b2 ----
        gemm128_kernel<EPI_GELU_Q_BF16><<<SK / 16, 256, 0, stream>>>(bfA, Wc1t, nullptr, qbuf, nullptr, bfB, qdiv);
        gemm128_kernel<EPI_RAW_F32><<<SK / 16, 256, 0, stream>>>(bfB, Wc2, cat_b2 + i * HDIM, nullptr, zbuf, nullptr, 1);

        // ---- residual + LayerNorm + valid mask (in-place h) ----
        ln_res_kernel<<<(SK * 32 + 255) / 256, 256, 0, stream>>>(
            zbuf, ln_g + i * HDIM, ln_b + i * HDIM, valid, h, SK);
    }

    // ---- final pooling into d_out [NG,128] ----
    hipMemsetAsync(d_out, 0, (size_t)out_size * 4, stream);
    pool_out_kernel<<<Nn, HDIM, 0, stream>>>(h, valid, batch, (float*)d_out, m, kk);
}
